// SupervisedContrastiveLoss_57234734186568
// MI455X (gfx1250) — compile-verified, matrix operand order checked by reading, not run on hardware
//
#include <hip/hip_runtime.h>
#include <hip/hip_bf16.h>

typedef __attribute__((ext_vector_type(16))) __bf16 v16bf;
typedef __attribute__((ext_vector_type(8)))  __bf16 v8bf;
typedef __attribute__((ext_vector_type(8)))  float  v8f;

#define TEMP_INV 10.0f   // 1 / TEMPERATURE
#define EPS 1e-8f

// ---------------------------------------------------------------------------
// Load a 16x32 bf16 fragment (M rows x K cols, row-major, row stride = ldk
// elements) in the CDNA5 WMMA A/B register layout:
//   lane 0-15 : M = lane,    elems 0-7 -> K = kBase+0..7,  elems 8-15 -> K = kBase+16..23
//   lane 16-31: M = lane-16, elems 0-7 -> K = kBase+8..15, elems 8-15 -> K = kBase+24..31
// Each half is a contiguous 16-byte chunk -> two b128 loads per lane.
// ---------------------------------------------------------------------------
template <typename PT>
__device__ inline v16bf load_frag16x32(PT base, int ldk, int kBase, int lane) {
    const int half = (lane >> 4) & 1;
    const int m    = lane & 15;
    PT p = base + (size_t)m * ldk + kBase + half * 8;
    v8bf lo = *reinterpret_cast<const v8bf*>(p);
    v8bf hi = *reinterpret_cast<const v8bf*>(p + 16);
    v16bf f;
#pragma unroll
    for (int i = 0; i < 8; ++i) { f[i] = lo[i]; f[i + 8] = hi[i]; }
    return f;
}

// ---------------------------------------------------------------------------
// Kernel 0: zero the scalar output (harness poisons d_out).
// ---------------------------------------------------------------------------
__global__ void scl_zero_out(float* out) { out[0] = 0.0f; }

// ---------------------------------------------------------------------------
// Kernel 1: row-normalize X (cosine prep) and emit bf16. One block per row.
// ---------------------------------------------------------------------------
__global__ void scl_normalize(const float* __restrict__ X,
                              __bf16* __restrict__ rn, int D) {
    const int row = blockIdx.x;
    const float* x = X + (size_t)row * D;
    float ss = 0.0f;
    for (int k = threadIdx.x; k < D; k += blockDim.x) {
        float v = x[k];
        ss += v * v;
    }
#pragma unroll
    for (int m = 16; m >= 1; m >>= 1) ss += __shfl_xor(ss, m, 32);
    __shared__ float red[8];
    const int lane = threadIdx.x & 31;
    const int w    = threadIdx.x >> 5;
    if (lane == 0) red[w] = ss;
    __syncthreads();
    if (threadIdx.x == 0) {
        float t = 0.0f;
        const int nw = blockDim.x >> 5;
        for (int i = 0; i < nw; ++i) t += red[i];
        red[0] = t;
    }
    __syncthreads();
    const float rinv = 1.0f / fmaxf(sqrtf(red[0]), EPS);
    for (int k = threadIdx.x; k < D; k += blockDim.x)
        rn[(size_t)row * D + k] = (__bf16)(x[k] * rinv);
}

// ---------------------------------------------------------------------------
// Kernel 2: convert soft labels to bf16.
// ---------------------------------------------------------------------------
__global__ void scl_cvt_soft(const float* __restrict__ S,
                             __bf16* __restrict__ sb, int n) {
    int i = blockIdx.x * blockDim.x + threadIdx.x;
    if (i < n) sb[i] = (__bf16)S[i];
}

// ---------------------------------------------------------------------------
// Kernel 3: fused contrastive loss. One block per 16-row strip, 8 waves.
// Each wave owns 16x64 column blocks (4 accumulators): per K-step one
// A-fragment (LDS) feeds 4 v_wmma_f32_16x16x32_bf16. Diagonal terms are
// handled by an exact correction in the single diagonal tile (uniform
// branch) instead of per-element predicates.
// ---------------------------------------------------------------------------
__global__ void __launch_bounds__(256)
scl_main(const __bf16* __restrict__ rn, const __bf16* __restrict__ sb,
         const int* __restrict__ labels, const int* __restrict__ pwPtr,
         float* __restrict__ out, int B, int D, int C) {
    __shared__ __align__(16) __bf16 ldsA[16 * 1024];   // 32 KB row panel
    __shared__ float ldsDen[16];
    __shared__ float ldsRowW[16];
    __shared__ float ldsWZ;
    __shared__ float ldsLogSum;

    const int i0 = blockIdx.x * 16;

    {
        const uint4* src = reinterpret_cast<const uint4*>(rn + (size_t)i0 * D);
        uint4* dst = reinterpret_cast<uint4*>(ldsA);
        const int n16 = (16 * D) / 8;
        for (int i = threadIdx.x; i < n16; i += blockDim.x) dst[i] = src[i];
    }
    if (threadIdx.x < 16) { ldsDen[threadIdx.x] = 0.0f; ldsRowW[threadIdx.x] = 0.0f; }
    if (threadIdx.x == 0) { ldsWZ = 0.0f; ldsLogSum = 0.0f; }
    __syncthreads();

    const int lane = threadIdx.x & 31;
    const int wave = threadIdx.x >> 5;           // 0..7
    const int half = lane >> 4;                  // C/D layout: M offset 0 or 8
    const int n    = lane & 15;                  // C/D layout: N = lane & 15
    const float pw = (float)pwPtr[0];

    int labRow[8];
#pragma unroll
    for (int r = 0; r < 8; ++r) labRow[r] = labels[i0 + r + half * 8];

    const v16bf as = load_frag16x32(sb + (size_t)i0 * C, C, 0, lane);

    float denAcc[8], rowWAcc[8];
#pragma unroll
    for (int r = 0; r < 8; ++r) { denAcc[r] = 0.0f; rowWAcc[r] = 0.0f; }
    float wz = 0.0f;

    const int nColBlk = B / 64;                  // 16x64 column blocks
    for (int cb = wave; cb < nColBlk; cb += 8) {
        const int j0 = cb * 64;

        // Warm L2/WGP$ for the NEXT column block (global_prefetch_b8).
        const int cbn = cb + 8;
        if (cbn < nColBlk) {
#pragma unroll
            for (int t = 0; t < 4; ++t)
                __builtin_prefetch(rn + (size_t)(cbn * 64 + t * 16) * D +
                                       (size_t)(lane & 15) * D, 0, 1);
        }

        // ---- Z: 16x64 = A(16xD) x B(64xD)^T, 4 WMMAs per A-fragment ----
        v8f acc0 = {}, acc1 = {}, acc2 = {}, acc3 = {};
        const __bf16* bp0 = rn + (size_t)(j0 +  0) * D;
        const __bf16* bp1 = rn + (size_t)(j0 + 16) * D;
        const __bf16* bp2 = rn + (size_t)(j0 + 32) * D;
        const __bf16* bp3 = rn + (size_t)(j0 + 48) * D;
        for (int k = 0; k < D; k += 32) {
            v16bf b0 = load_frag16x32(bp0, D, k, lane);
            v16bf b1 = load_frag16x32(bp1, D, k, lane);
            v16bf b2 = load_frag16x32(bp2, D, k, lane);
            v16bf b3 = load_frag16x32(bp3, D, k, lane);
            v16bf a  = load_frag16x32(ldsA, D, k, lane);     // LDS, reused 4x
            acc0 = __builtin_amdgcn_wmma_f32_16x16x32_bf16(
                false, a, false, b0, (short)0, acc0, false, false);
            acc1 = __builtin_amdgcn_wmma_f32_16x16x32_bf16(
                false, a, false, b1, (short)0, acc1, false, false);
            acc2 = __builtin_amdgcn_wmma_f32_16x16x32_bf16(
                false, a, false, b2, (short)0, acc2, false, false);
            acc3 = __builtin_amdgcn_wmma_f32_16x16x32_bf16(
                false, a, false, b3, (short)0, acc3, false, false);
        }

        // ---- epilogue per 16x16 tile ----
#pragma unroll
        for (int t = 0; t < 4; ++t) {
            const v8f accT = (t == 0) ? acc0 : (t == 1) ? acc1
                            : (t == 2) ? acc2 : acc3;
            const int jt0 = j0 + t * 16;

            v16bf bs = load_frag16x32(sb + (size_t)jt0 * C, C, 0, lane);
            v8f soft = {};
            soft = __builtin_amdgcn_wmma_f32_16x16x32_bf16(
                false, as, false, bs, (short)0, soft, false, false);

            const int lj = labels[jt0 + n];
#pragma unroll
            for (int r = 0; r < 8; ++r) {
                const float z = accT[r] * TEMP_INV;
                float w = pw * soft[r] + ((labRow[r] == lj) ? 1.0f : 0.0f);
                wz         += w * z;
                rowWAcc[r] += w;
                denAcc[r]  += __expf(z);          // includes exp(Z_ii)
            }

            if (jt0 == i0) {                      // uniform branch: diag tile
#pragma unroll
                for (int r = 0; r < 8; ++r) {
                    if (n == r + half * 8) {      // the (m == n) element
                        const float zd = accT[r] * TEMP_INV;
                        denAcc[r]  -= __expf(zd); // denom excludes j == i
                        rowWAcc[r] -= 1.0f;       // pos excludes j == i
                        wz         -= zd;
                    }
                }
            }
        }
    }

#pragma unroll
    for (int m = 8; m >= 1; m >>= 1) {
#pragma unroll
        for (int r = 0; r < 8; ++r) {
            denAcc[r]  += __shfl_xor(denAcc[r],  m, 32);
            rowWAcc[r] += __shfl_xor(rowWAcc[r], m, 32);
        }
    }
    if (n == 0) {
#pragma unroll
        for (int r = 0; r < 8; ++r) {
            atomicAdd(&ldsDen[r + half * 8],  denAcc[r]);
            atomicAdd(&ldsRowW[r + half * 8], rowWAcc[r]);
        }
    }
#pragma unroll
    for (int m = 16; m >= 1; m >>= 1) wz += __shfl_xor(wz, m, 32);
    if (lane == 0) atomicAdd(&ldsWZ, wz);
    __syncthreads();

    if (threadIdx.x < 16) {
        const float c = ldsRowW[threadIdx.x] * __logf(ldsDen[threadIdx.x]);
        atomicAdd(&ldsLogSum, c);
    }
    __syncthreads();
    if (threadIdx.x == 0)
        atomicAdd(out, (ldsLogSum - ldsWZ) / (float)B);
}

// ---------------------------------------------------------------------------
extern "C" void kernel_launch(void* const* d_in, const int* in_sizes, int n_in,
                              void* d_out, int out_size, void* d_ws, size_t ws_size,
                              hipStream_t stream) {
    const float* X      = (const float*)d_in[0];   // representations [B, D]
    const float* S      = (const float*)d_in[1];   // soft_labels     [B, C]
    const int*   labels = (const int*)d_in[2];     // labels          [B]
    const int*   pw     = (const int*)d_in[3];     // pseudo_weight scalar

    const int B = in_sizes[2];
    const int D = in_sizes[0] / B;
    const int C = in_sizes[1] / B;

    __bf16* rn = (__bf16*)d_ws;                    // [B, D] bf16 (8 MB)
    __bf16* sb = rn + (size_t)B * D;               // [B, C] bf16 (256 KB)

    scl_zero_out<<<1, 1, 0, stream>>>((float*)d_out);
    scl_normalize<<<B, 256, 0, stream>>>(X, rn, D);
    scl_cvt_soft<<<(B * C + 255) / 256, 256, 0, stream>>>(S, sb, B * C);
    scl_main<<<B / 16, 256, 0, stream>>>(rn, sb, labels, pw,
                                         (float*)d_out, B, D, C);
}
